// PruneRNN_26817775796426
// MI455X (gfx1250) — compile-verified, matrix operand order checked by reading, not run on hardware
//
#include <hip/hip_runtime.h>
#include <hip/hip_bf16.h>

#define SEQ 2048
#define BATCH 64
#define INF 256
#define HID 512
#define SCAN_NWG 8

typedef __attribute__((ext_vector_type(16))) __bf16 v16bf;
typedef __attribute__((ext_vector_type(8)))  float  v8f;

// ---------------------------------------------------------------------------
// CDNA5 async global->LDS copy (ASYNCcnt-tracked), with portable fallback.
// Probe: builtin's param0 is 'int __vector(4) __device__ *' (AS1 v4i*), so
// feed it v4i pointers in AS1/AS3. LDS aperture rule (ISA 10.2): generic LDS
// address low 32 bits == LDS offset, so integer casts produce valid pointers.
// ---------------------------------------------------------------------------
#if defined(__HIP_DEVICE_COMPILE__) && \
    __has_builtin(__builtin_amdgcn_global_load_async_to_lds_b128) && \
    __has_builtin(__builtin_amdgcn_s_wait_asynccnt)
#define USE_ASYNC_LDS 1
#else
#define USE_ASYNC_LDS 0
#endif

#if USE_ASYNC_LDS
typedef int v4i_ __attribute__((vector_size(16)));
typedef __attribute__((address_space(1))) v4i_ g1_v4i;
typedef __attribute__((address_space(3))) v4i_ l3_v4i;
__device__ __forceinline__ void async_cp16(const void* g, void* l) {
    __builtin_amdgcn_global_load_async_to_lds_b128(
        (g1_v4i*)(size_t)g, (l3_v4i*)(unsigned)(size_t)l, 0, 0);
}
__device__ __forceinline__ void async_fence() {
    __builtin_amdgcn_s_wait_asynccnt(0);
}
#else
__device__ __forceinline__ void async_cp16(const void* g, void* l) {
    *(uint4*)l = *(const uint4*)g;
}
__device__ __forceinline__ void async_fence() {}
#endif

// ---------------------------------------------------------------------------
// WMMA helpers
// ---------------------------------------------------------------------------
__device__ __forceinline__ v8f wmma_bf16(v16bf a, v16bf b, v8f c) {
    // D = A(16x32) * B(32x16) + C(16x16), f32 accumulate
    return __builtin_amdgcn_wmma_f32_16x16x32_bf16(
        false, a, false, b, (short)0, c, false, false);
}

// Load one 16x32 bf16 fragment from an LDS tile row (row-major, K contiguous).
// Lane half (l>>4) selects +8 K offset; elems 0..7 = K [c0,c0+8),
// elems 8..15 = K [c0+16,c0+24).
struct Frag128x2 { uint4 lo; uint4 hi; };
__device__ __forceinline__ v16bf frag_ld(const __bf16* row, int c0) {
    Frag128x2 p;
    p.lo = *(const uint4*)(row + c0);
    p.hi = *(const uint4*)(row + c0 + 16);
    return __builtin_bit_cast(v16bf, p);
}

// ---------------------------------------------------------------------------
// Grid barrier (8 co-resident WGs)
// ---------------------------------------------------------------------------
__device__ __forceinline__ void grid_barrier(unsigned* bar, int nwg) {
    __syncthreads();
    if (threadIdx.x == 0) {
        __threadfence();
        unsigned* cnt = bar;
        unsigned* gen = bar + 1;
        unsigned g = __atomic_load_n(gen, __ATOMIC_ACQUIRE);
        if (atomicAdd(cnt, 1u) == (unsigned)(nwg - 1)) {
            __atomic_store_n(cnt, 0u, __ATOMIC_RELAXED);
            __atomic_fetch_add(gen, 1u, __ATOMIC_RELEASE);
        } else {
            while (__atomic_load_n(gen, __ATOMIC_ACQUIRE) == g) {
                __builtin_amdgcn_s_sleep(1);
            }
        }
    }
    __syncthreads();
}

__global__ void init_barrier_k(unsigned* bar) { bar[0] = 0u; bar[1] = 0u; }

// ---------------------------------------------------------------------------
// Weight convert: fp32 * mask -> bf16 (masks are jnp.bool_ = 1 byte)
// ---------------------------------------------------------------------------
__global__ void cvt_mask_k(const float* __restrict__ w,
                           const unsigned char* __restrict__ m,
                           __bf16* __restrict__ o, int n) {
    int i = blockIdx.x * 256 + threadIdx.x;
    if (i < n) o[i] = (__bf16)(m[i] ? w[i] : 0.0f);
}

// ---------------------------------------------------------------------------
// Input projection GEMM: out[M,N] = A[M,K] * W[N,K]^T + bias1[N] + bias2[N]
// WG: 128 threads = 4 waves. WG tile 128x128, wave tile 64x64 (4x4 WMMA accs).
// A is fp32 (layer 0: x) or bf16 (layer 1: ys0); W is pre-converted bf16.
// ---------------------------------------------------------------------------
#define LSTR 40   // LDS row stride in bf16 elems (32 data + 8 pad)

template <bool ABF16>
__global__ __launch_bounds__(128)
void proj_gemm_k(const void* __restrict__ Av, const __bf16* __restrict__ W,
                 const float* __restrict__ bias1, const float* __restrict__ bias2,
                 float* __restrict__ out, int M, int N, int K) {
    __shared__ __bf16 As[128 * LSTR];
    __shared__ __bf16 Bs[128 * LSTR];

    const int tid  = threadIdx.x;
    const int lane = tid & 31;
    const int wv   = tid >> 5;          // wave 0..3
    const int half = lane >> 4;
    const int l16  = lane & 15;
    const int m0   = blockIdx.x * 128;
    const int n0   = blockIdx.y * 128;
    const int wm   = (wv & 1) * 64;     // wave M offset in tile
    const int wn   = (wv >> 1) * 64;    // wave N offset in tile

    v8f acc[4][4] = {};

    for (int k0 = 0; k0 < K; k0 += 32) {
        // --- stage A row `tid` (32 elems) ---
        if constexpr (ABF16) {
            const __bf16* src = (const __bf16*)Av + (size_t)(m0 + tid) * K + k0;
            __bf16* dst = &As[tid * LSTR];
            async_cp16(src,      dst);
            async_cp16(src + 8,  dst + 8);
            async_cp16(src + 16, dst + 16);
            async_cp16(src + 24, dst + 24);
        } else {
            const float* src = (const float*)Av + (size_t)(m0 + tid) * K + k0;
            __bf16* dst = &As[tid * LSTR];
#pragma unroll
            for (int i = 0; i < 32; i += 4) {
                float4 v = *(const float4*)(src + i);
                dst[i + 0] = (__bf16)v.x; dst[i + 1] = (__bf16)v.y;
                dst[i + 2] = (__bf16)v.z; dst[i + 3] = (__bf16)v.w;
            }
        }
        // --- stage W row `tid` (32 elems, already bf16) ---
        {
            const __bf16* src = W + (size_t)(n0 + tid) * K + k0;
            __bf16* dst = &Bs[tid * LSTR];
            async_cp16(src,      dst);
            async_cp16(src + 8,  dst + 8);
            async_cp16(src + 16, dst + 16);
            async_cp16(src + 24, dst + 24);
        }
        async_fence();
        __syncthreads();

        v16bf af[4], bf[4];
#pragma unroll
        for (int i = 0; i < 4; i++) {
            af[i] = frag_ld(&As[(wm + i * 16 + l16) * LSTR], half * 8);
            bf[i] = frag_ld(&Bs[(wn + i * 16 + l16) * LSTR], half * 8);
        }
#pragma unroll
        for (int mi = 0; mi < 4; mi++)
#pragma unroll
            for (int ni = 0; ni < 4; ni++)
                acc[mi][ni] = wmma_bf16(af[mi], bf[ni], acc[mi][ni]);
        __syncthreads();
    }

    // epilogue: D vgpr j -> M = half*8 + j, N = l16
#pragma unroll
    for (int ni = 0; ni < 4; ni++) {
        int n = n0 + wn + ni * 16 + l16;
        float b = bias1[n] + bias2[n];
#pragma unroll
        for (int mi = 0; mi < 4; mi++) {
            int mb = m0 + wm + mi * 16 + half * 8;
#pragma unroll
            for (int j = 0; j < 8; j++) {
                out[(size_t)(mb + j) * N + n] = acc[mi][ni][j] + b;
            }
        }
    }
}

// ---------------------------------------------------------------------------
// Recurrent scan: h[t] = tanh(xw[t] + h[t-1] @ Whh^T), 8 persistent WGs.
// WG g owns columns [g*64, g*64+64). Whh slice lives in LDS for whole scan.
// h state ping-pongs in global (bf16), one grid barrier per step.
// WG: 256 threads = 8 waves, wave tile 16(M) x 32(N) -> 2 WMMA accs,
// 16 K-chunks -> 32 wmma per wave per step.
// ---------------------------------------------------------------------------
#define HSTR 520  // LDS row stride for 512-wide rows (+8 pad)

__global__ __launch_bounds__(256)
void rnn_scan_k(const float* __restrict__ xw, const __bf16* __restrict__ whh,
                __bf16* __restrict__ hbuf, __bf16* __restrict__ ys,
                float* __restrict__ fin, int S, unsigned* bar) {
    extern __shared__ __align__(16) char smem[];
    __bf16* whh_s = (__bf16*)smem;            // [64][HSTR]
    __bf16* h_s   = whh_s + 64 * HSTR;        // [64][HSTR]

    const int tid   = threadIdx.x;
    const int wg    = blockIdx.x;             // 0..7
    const int lane  = tid & 31;
    const int wv    = tid >> 5;               // 0..7
    const int half  = lane >> 4;
    const int l16   = lane & 15;
    const int m0    = (wv & 3) * 16;          // batch rows m0..m0+15
    const int n0l   = (wv >> 2) * 32;         // local col offset 0 or 32
    const int nbase = wg * 64;

    // Load this WG's Whh slice into LDS once (rows nbase..nbase+63, K=512).
    {
        int r = tid >> 2, q = tid & 3;        // 4 threads per row, 256B each
        const __bf16* src = whh + (size_t)(nbase + r) * HID + q * 128;
        __bf16* dst = whh_s + r * HSTR + q * 128;
#pragma unroll
        for (int i = 0; i < 16; i++) async_cp16(src + i * 8, dst + i * 8);
    }
    // Zero our slice of h ping buffer 0 (initial hidden state).
    for (int i = tid; i < 64 * 64; i += 256) {
        int r = i >> 6, c = i & 63;
        hbuf[(size_t)r * HID + nbase + c] = (__bf16)0.0f;
    }
    async_fence();
    __threadfence();
    grid_barrier(bar, SCAN_NWG);

    for (int t = 0; t < S; t++) {
        const __bf16* hcur = hbuf + (size_t)(t & 1) * BATCH * HID;
        __bf16*       hnxt = hbuf + (size_t)((t + 1) & 1) * BATCH * HID;

        // Stage full h[t-1] (64x512 bf16) into LDS via async-to-LDS
        // (all 8 WGs read the same 64KB -> L2 broadcast).
        {
            int r = tid >> 2, q = tid & 3;
            const __bf16* src = hcur + (size_t)r * HID + q * 128;
            __bf16* dst = h_s + r * HSTR + q * 128;
#pragma unroll
            for (int i = 0; i < 16; i++) async_cp16(src + i * 8, dst + i * 8);
        }
        // Warm L2 with next step's xw tile while this step computes.
        if (t + 1 < S) {
            __builtin_prefetch(xw + (size_t)(t + 1) * BATCH * HID + (size_t)tid * 128, 0, 1);
        }
        async_fence();
        __syncthreads();

        v8f acc[2] = {};
#pragma unroll
        for (int kc = 0; kc < 16; kc++) {
            int k0 = kc * 32;
            v16bf a  = frag_ld(&h_s[(m0 + l16) * HSTR + k0], half * 8);
            v16bf b0 = frag_ld(&whh_s[(n0l + l16) * HSTR + k0], half * 8);
            v16bf b1 = frag_ld(&whh_s[(n0l + 16 + l16) * HSTR + k0], half * 8);
            acc[0] = wmma_bf16(a, b0, acc[0]);
            acc[1] = wmma_bf16(a, b1, acc[1]);
        }

        const float* xwt = xw + (size_t)t * BATCH * HID;
        __bf16* yst = ys ? (ys + (size_t)t * BATCH * HID) : nullptr;
#pragma unroll
        for (int ni = 0; ni < 2; ni++) {
            int n = nbase + n0l + ni * 16 + l16;
#pragma unroll
            for (int j = 0; j < 8; j++) {
                int m = m0 + half * 8 + j;
                float v = tanhf(acc[ni][j] + xwt[(size_t)m * HID + n]);
                __bf16 hv = (__bf16)v;
                hnxt[(size_t)m * HID + n] = hv;
                if (yst) yst[(size_t)m * HID + n] = hv;
                if (fin && t == S - 1) fin[(size_t)m * HID + n] = v;
            }
        }
        __threadfence();
        grid_barrier(bar, SCAN_NWG);  // h[t] fully published; safe to advance
    }
}

// ---------------------------------------------------------------------------
// Launch
// ---------------------------------------------------------------------------
extern "C" void kernel_launch(void* const* d_in, const int* in_sizes, int n_in,
                              void* d_out, int out_size, void* d_ws, size_t ws_size,
                              hipStream_t stream) {
    const float* x    = (const float*)d_in[0];
    const float* Wih0 = (const float*)d_in[1];
    const float* Whh0 = (const float*)d_in[2];
    const float* bih0 = (const float*)d_in[3];
    const float* bhh0 = (const float*)d_in[4];
    const float* Wih1 = (const float*)d_in[5];
    const float* Whh1 = (const float*)d_in[6];
    const float* bih1 = (const float*)d_in[7];
    const float* bhh1 = (const float*)d_in[8];
    const unsigned char* mih0 = (const unsigned char*)d_in[9];
    const unsigned char* mhh0 = (const unsigned char*)d_in[10];
    const unsigned char* mih1 = (const unsigned char*)d_in[11];
    const unsigned char* mhh1 = (const unsigned char*)d_in[12];

    char* p = (char*)d_ws;
    float*  xw    = (float*)p;  p += (size_t)SEQ * BATCH * HID * sizeof(float);   // 256 MiB
    __bf16* ys0   = (__bf16*)p; p += (size_t)SEQ * BATCH * HID * sizeof(__bf16);  // 128 MiB
    __bf16* wih0b = (__bf16*)p; p += (size_t)HID * INF * sizeof(__bf16);
    __bf16* whh0b = (__bf16*)p; p += (size_t)HID * HID * sizeof(__bf16);
    __bf16* wih1b = (__bf16*)p; p += (size_t)HID * HID * sizeof(__bf16);
    __bf16* whh1b = (__bf16*)p; p += (size_t)HID * HID * sizeof(__bf16);
    __bf16* hbuf  = (__bf16*)p; p += (size_t)2 * BATCH * HID * sizeof(__bf16);
    unsigned* bar = (unsigned*)p;

    init_barrier_k<<<1, 1, 0, stream>>>(bar);

    cvt_mask_k<<<(HID * INF + 255) / 256, 256, 0, stream>>>(Wih0, mih0, wih0b, HID * INF);
    cvt_mask_k<<<(HID * HID + 255) / 256, 256, 0, stream>>>(Whh0, mhh0, whh0b, HID * HID);
    cvt_mask_k<<<(HID * HID + 255) / 256, 256, 0, stream>>>(Wih1, mih1, wih1b, HID * HID);
    cvt_mask_k<<<(HID * HID + 255) / 256, 256, 0, stream>>>(Whh1, mhh1, whh1b, HID * HID);

    const int M = SEQ * BATCH;           // 131072
    dim3 grid(M / 128, HID / 128);       // 1024 x 4

    // Layer 0 input projection: xw0 = x @ Wih0^T + bih0 + bhh0
    proj_gemm_k<false><<<grid, 128, 0, stream>>>(x, wih0b, bih0, bhh0, xw, M, HID, INF);

    size_t smem = (size_t)2 * 64 * HSTR * sizeof(__bf16);  // 130 KiB
    // Layer 0 scan -> ys0 (bf16)
    rnn_scan_k<<<SCAN_NWG, 256, smem, stream>>>(xw, whh0b, hbuf, ys0, nullptr, SEQ, bar);

    // Layer 1 input projection: xw1 = ys0 @ Wih1^T + bih1 + bhh1 (reuses xw buffer)
    proj_gemm_k<true><<<grid, 128, 0, stream>>>(ys0, wih1b, bih1, bhh1, xw, M, HID, HID);

    // Layer 1 scan -> final hidden state to d_out
    rnn_scan_k<<<SCAN_NWG, 256, smem, stream>>>(xw, whh1b, hbuf, nullptr, (float*)d_out, SEQ, bar);
}
